// FullLanguageZone_7249904796043
// MI455X (gfx1250) — matrix operand order repaired and checked by reading, not compiled, not run on hardware
//
#include <hip/hip_runtime.h>
#include <hip/hip_bf16.h>

typedef __attribute__((ext_vector_type(16))) _Float16 v16h;
typedef __attribute__((ext_vector_type(8)))  _Float16 v8h;
typedef __attribute__((ext_vector_type(8)))  float    v8f;

#define N_TOK   8192
#define D_MODEL 1024
#define HIDDEN  2048
#define MOE_DIM 64
#define N_EXP   8
#define EXP_HID 1024

#define BK 32

// ---------------------------------------------------------------------------
// f32 -> f16 elementwise conversion
// ---------------------------------------------------------------------------
__global__ void f2h_kernel(const float* __restrict__ in, _Float16* __restrict__ out, long n) {
    long i = (long)blockIdx.x * blockDim.x + threadIdx.x;
    if (i < n) out[i] = (_Float16)in[i];
}

// ---------------------------------------------------------------------------
// Tiled WMMA GEMM: D = act(A[MxK] @ B[KxN] + bias[N])
//   ACT: 0=none, 1=sigmoid, 2=relu
//   EPI: 0 -> out16[m*N+n]  = (f16)v
//        1 -> outf[m*64+n] += comb[m*8+eidx] * v
//        2 -> outf[m*64+n]  = comb[m*8+eidx] * v
// Geometry: 256 threads = 8 waves, BM = WAVE_M*MT*16 (=128), BN = WAVE_N*NT*16.
// Requirements: M % BM == 0, K % 32 == 0, N % BN == 0 (no guards in hot loop).
// Register-double-buffered global->LDS staging; fragments per ISA wave32
// 16-bit A layout (M row per lane, K split across lane halves) and B layout
// (N col = lane%16, K split across lane halves).
// ---------------------------------------------------------------------------
template <int ACT, int EPI, int WAVE_M, int WAVE_N, int MT, int NT>
__global__ __launch_bounds__(256)
void gemm_kernel(const _Float16* __restrict__ A, const _Float16* __restrict__ B,
                 const float* __restrict__ bias,
                 _Float16* __restrict__ out16, float* __restrict__ outf,
                 const float* __restrict__ comb, int eidx,
                 int M, int N, int K) {
    constexpr int BM_  = WAVE_M * MT * 16;               // 128
    constexpr int BN_  = WAVE_N * NT * 16;               // 128 or 64
    constexpr int ASEG = (BM_ * BK) / (256 * 8);         // v8h segments per thread (A)
    constexpr int BSEG = (BK * BN_) / (256 * 8);         // v8h segments per thread (B)

    __shared__ _Float16 As[BM_][BK];
    __shared__ _Float16 Bs[BK][BN_];

    const int tid  = threadIdx.x;
    const int lane = tid & 31;
    const int wave = tid >> 5;
    const int wm   = wave / WAVE_N;
    const int wn   = wave % WAVE_N;
    const int bm   = blockIdx.y * BM_;
    const int bn   = blockIdx.x * BN_;

    const int half16 = lane >> 4;

    v8f acc[MT][NT];
#pragma unroll
    for (int i = 0; i < MT; ++i)
#pragma unroll
        for (int j = 0; j < NT; ++j)
#pragma unroll
            for (int r = 0; r < 8; ++r) acc[i][j][r] = 0.f;

    // staging registers (double buffer in regs)
    v8h ar[ASEG], br[BSEG];

    auto load_tile = [&](int k0) {
#pragma unroll
        for (int s = 0; s < ASEG; ++s) {
            int idx = (s * 256 + tid) * 8;
            int r = idx / BK, c = idx % BK;
            ar[s] = *(const v8h*)(A + (long)(bm + r) * K + (k0 + c));
        }
#pragma unroll
        for (int s = 0; s < BSEG; ++s) {
            int idx = (s * 256 + tid) * 8;
            int r = idx / BN_, c = idx % BN_;
            br[s] = *(const v8h*)(B + (long)(k0 + r) * N + (bn + c));
        }
    };

    load_tile(0);

    for (int k0 = 0; k0 < K; k0 += BK) {
        // commit staged registers to LDS
#pragma unroll
        for (int s = 0; s < ASEG; ++s) {
            int idx = (s * 256 + tid) * 8;
            *(v8h*)&As[idx / BK][idx % BK] = ar[s];
        }
#pragma unroll
        for (int s = 0; s < BSEG; ++s) {
            int idx = (s * 256 + tid) * 8;
            *(v8h*)&Bs[idx / BN_][idx % BN_] = br[s];
        }
        __syncthreads();

        // overlap: next tile's global loads issue while WMMAs consume LDS
        if (k0 + BK < K) load_tile(k0 + BK);
        if (k0 + 2 * BK < K) {
            __builtin_prefetch(A + (long)(bm + tid / 2) * K + (k0 + 2 * BK), 0, 3);
            __builtin_prefetch(B + (long)(k0 + 2 * BK + (tid >> 3)) * N + bn, 0, 3);
        }

        v16h afrag[MT], bfrag[NT];
        const int ak = half16 * 8;
#pragma unroll
        for (int i = 0; i < MT; ++i) {
            int r = wm * (MT * 16) + i * 16 + (lane & 15);
            v8h lo = *(const v8h*)&As[r][ak];        // K = half*8 .. +7
            v8h hi = *(const v8h*)&As[r][16 + ak];   // K = 16+half*8 .. +7
#pragma unroll
            for (int j = 0; j < 8; ++j) { afrag[i][j] = lo[j]; afrag[i][8 + j] = hi[j]; }
        }
#pragma unroll
        for (int jn = 0; jn < NT; ++jn) {
            int c = wn * (NT * 16) + jn * 16;
            int cn = c + (lane & 15);                // N col = lane%16, K = half*8.. and 16+half*8..
#pragma unroll
            for (int j = 0; j < 8; ++j) { bfrag[jn][j] = Bs[ak + j][cn]; bfrag[jn][8 + j] = Bs[16 + ak + j][cn]; }
        }

#pragma unroll
        for (int i = 0; i < MT; ++i)
#pragma unroll
            for (int j = 0; j < NT; ++j)
                acc[i][j] = __builtin_amdgcn_wmma_f32_16x16x32_f16(
                    false, afrag[i], false, bfrag[j], (short)0, acc[i][j], false, false);

        __syncthreads();
    }

    // epilogue (C layout: VGPR r -> M = r + 8*half, N = lane%16)
#pragma unroll
    for (int i = 0; i < MT; ++i) {
#pragma unroll
        for (int j = 0; j < NT; ++j) {
            int n = bn + wn * (NT * 16) + j * 16 + (lane & 15);
            float bv = bias[n];
            int mbase = bm + wm * (MT * 16) + i * 16 + half16 * 8;
#pragma unroll
            for (int r = 0; r < 8; ++r) {
                int m = mbase + r;
                float v = acc[i][j][r] + bv;
                if (ACT == 1) v = 1.f / (1.f + __expf(-v));
                if (ACT == 2) v = fmaxf(v, 0.f);
                if (EPI == 0) {
                    out16[(long)m * N + n] = (_Float16)v;
                } else {
                    float w = comb[m * N_EXP + eidx];
                    long  o = (long)m * MOE_DIM + n;
                    if (EPI == 1) outf[o] += w * v;
                    else          outf[o]  = w * v;
                }
            }
        }
    }
}

// ---------------------------------------------------------------------------
// Router: rh = relu(cont @ r_w1 + b1); probs = softmax(rh @ r_w2 + b2)
// top-2 renormalized -> comb[N_TOK, 8]
// ---------------------------------------------------------------------------
__global__ __launch_bounds__(256)
void router_kernel(const _Float16* __restrict__ cont, const float* __restrict__ w1,
                   const float* __restrict__ b1, const float* __restrict__ w2,
                   const float* __restrict__ b2, float* __restrict__ comb, int ntok) {
    int n = blockIdx.x * blockDim.x + threadIdx.x;
    if (n >= ntok) return;
    float x[MOE_DIM];
#pragma unroll
    for (int d = 0; d < MOE_DIM; ++d) x[d] = (float)cont[(long)n * MOE_DIM + d];
    float logits[N_EXP];
#pragma unroll
    for (int e = 0; e < N_EXP; ++e) logits[e] = b2[e];
    for (int h = 0; h < 64; ++h) {
        float s = b1[h];
        for (int d = 0; d < MOE_DIM; ++d) s += x[d] * w1[d * 64 + h];
        s = fmaxf(s, 0.f);
#pragma unroll
        for (int e = 0; e < N_EXP; ++e) logits[e] += s * w2[h * N_EXP + e];
    }
    float mx = logits[0];
#pragma unroll
    for (int e = 1; e < N_EXP; ++e) mx = fmaxf(mx, logits[e]);
    float p[N_EXP], sum = 0.f;
#pragma unroll
    for (int e = 0; e < N_EXP; ++e) { p[e] = __expf(logits[e] - mx); sum += p[e]; }
#pragma unroll
    for (int e = 0; e < N_EXP; ++e) p[e] /= sum;
    int i1 = 0;
#pragma unroll
    for (int e = 1; e < N_EXP; ++e) if (p[e] > p[i1]) i1 = e;
    int i2 = (i1 == 0) ? 1 : 0;
#pragma unroll
    for (int e = 0; e < N_EXP; ++e) if (e != i1 && p[e] > p[i2]) i2 = e;
    float s2 = p[i1] + p[i2];
#pragma unroll
    for (int e = 0; e < N_EXP; ++e)
        comb[(long)n * N_EXP + e] = (e == i1) ? p[i1] / s2 : ((e == i2) ? p[i2] / s2 : 0.f);
}

// ---------------------------------------------------------------------------
// LayerNorm over D_MODEL per row (input = sigmoid'd decoder activations, f16)
// ---------------------------------------------------------------------------
__global__ __launch_bounds__(256)
void layernorm_kernel(const _Float16* __restrict__ dec, const float* __restrict__ g,
                      const float* __restrict__ b, float* __restrict__ out, int D) {
    int row = blockIdx.x;
    int tid = threadIdx.x;
    __shared__ float s1[256];
    __shared__ float s2[256];
    float lsum = 0.f, lsq = 0.f;
    for (int d = tid; d < D; d += 256) {
        float v = (float)dec[(long)row * D + d];
        lsum += v; lsq += v * v;
    }
    s1[tid] = lsum; s2[tid] = lsq;
    __syncthreads();
    for (int s = 128; s > 0; s >>= 1) {
        if (tid < s) { s1[tid] += s1[tid + s]; s2[tid] += s2[tid + s]; }
        __syncthreads();
    }
    float mu  = s1[0] / D;
    float var = s2[0] / D - mu * mu;
    float rs  = rsqrtf(var + 1e-5f);
    for (int d = tid; d < D; d += 256) {
        float v = (float)dec[(long)row * D + d];
        out[(long)row * D + d] = (v - mu) * rs * g[d] + b[d];
    }
}

// ---------------------------------------------------------------------------
// host-side orchestration
// ---------------------------------------------------------------------------
static inline void f2h(const float* in, _Float16* out, long n, hipStream_t s) {
    f2h_kernel<<<(unsigned)((n + 255) / 256), 256, 0, s>>>(in, out, n);
}

extern "C" void kernel_launch(void* const* d_in, const int* in_sizes, int n_in,
                              void* d_out, int out_size, void* d_ws, size_t ws_size,
                              hipStream_t stream) {
    (void)in_sizes; (void)n_in; (void)out_size; (void)ws_size;

    const float* X     = (const float*)d_in[0];
    const float* enc_w = (const float*)d_in[1];
    const float* enc_b = (const float*)d_in[2];
    const float* s2c_w = (const float*)d_in[3];
    const float* s2c_b = (const float*)d_in[4];
    const float* r_w1  = (const float*)d_in[5];
    const float* r_b1  = (const float*)d_in[6];
    const float* r_w2  = (const float*)d_in[7];
    const float* r_b2  = (const float*)d_in[8];
    const float* e_w1  = (const float*)d_in[9];
    const float* e_b1  = (const float*)d_in[10];
    const float* e_w2  = (const float*)d_in[11];
    const float* e_b2  = (const float*)d_in[12];
    const float* c2s_w = (const float*)d_in[13];
    const float* c2s_b = (const float*)d_in[14];
    const float* dec_w = (const float*)d_in[15];
    const float* dec_b = (const float*)d_in[16];
    const float* ln_g  = (const float*)d_in[17];
    const float* ln_b  = (const float*)d_in[18];

    constexpr size_t MBy = 1024ull * 1024ull;
    char* ws = (char*)d_ws;
    _Float16* X16     = (_Float16*)(ws + 0);                 // 16 MB (aliased as dec16)
    _Float16* dec16   = X16;
    _Float16* spk16   = (_Float16*)(ws + 16 * MBy);          // 32 MB (aliased H16/rates16)
    _Float16* H16     = spk16;
    _Float16* rates16 = spk16;
    _Float16* encW16  = (_Float16*)(ws + 48 * MBy);          // 4 MB
    _Float16* decW16  = (_Float16*)(ws + 52 * MBy);          // 4 MB
    _Float16* s2cW16  = (_Float16*)(ws + 56 * MBy);          // 256 KB
    _Float16* c2sW16  = (_Float16*)(ws + 56 * MBy + 256 * 1024);
    _Float16* eW1h    = (_Float16*)(ws + 57 * MBy);          // 1 MB
    _Float16* eW2h    = (_Float16*)(ws + 58 * MBy);          // 1 MB
    _Float16* cont16  = (_Float16*)(ws + 59 * MBy);          // 1 MB
    _Float16* moe16   = (_Float16*)(ws + 60 * MBy);          // 1 MB
    float*    comb    = (float*)   (ws + 61 * MBy);          // 256 KB
    float*    moeF    = (float*)   (ws + 61 * MBy + 256 * 1024); // 2 MB

    // f32 -> f16 conversions (weights + activations)
    f2h(X,     X16,    (long)N_TOK * D_MODEL,           stream);
    f2h(enc_w, encW16, (long)D_MODEL * HIDDEN,          stream);
    f2h(s2c_w, s2cW16, (long)HIDDEN * MOE_DIM,          stream);
    f2h(c2s_w, c2sW16, (long)MOE_DIM * HIDDEN,          stream);
    f2h(dec_w, decW16, (long)HIDDEN * D_MODEL,          stream);
    f2h(e_w1,  eW1h,   (long)N_EXP * MOE_DIM * EXP_HID, stream);
    f2h(e_w2,  eW2h,   (long)N_EXP * EXP_HID * MOE_DIM, stream);

    dim3 blk(256);
    // wide config: BM=128, BN=128 (N % 128 == 0)  -> <ACT,EPI,2,4,4,2>
    // narrow config: BM=128, BN=64 (N == 64)      -> <ACT,EPI,4,2,2,2>

    // 1) encoder: spikes = sigmoid(X @ enc_w + enc_b)   [8192 x 2048]
    gemm_kernel<1, 0, 2, 4, 4, 2><<<dim3(HIDDEN / 128, N_TOK / 128), blk, 0, stream>>>(
        X16, encW16, enc_b, spk16, nullptr, nullptr, 0, N_TOK, HIDDEN, D_MODEL);
    // 2) cont = spikes @ s2c_w + s2c_b                  [8192 x 64]
    gemm_kernel<0, 0, 4, 2, 2, 2><<<dim3(1, N_TOK / 128), blk, 0, stream>>>(
        spk16, s2cW16, s2c_b, cont16, nullptr, nullptr, 0, N_TOK, MOE_DIM, HIDDEN);
    // 3) router -> comb
    router_kernel<<<N_TOK / 256, blk, 0, stream>>>(cont16, r_w1, r_b1, r_w2, r_b2, comb, N_TOK);
    // 4) experts: moeF = sum_e comb[:,e] * (relu(cont@W1_e+b1_e)@W2_e + b2_e)
    for (int e = 0; e < N_EXP; ++e) {
        gemm_kernel<2, 0, 2, 4, 4, 2><<<dim3(EXP_HID / 128, N_TOK / 128), blk, 0, stream>>>(
            cont16, eW1h + (long)e * MOE_DIM * EXP_HID, e_b1 + (long)e * EXP_HID,
            H16, nullptr, nullptr, 0, N_TOK, EXP_HID, MOE_DIM);
        if (e == 0)
            gemm_kernel<0, 2, 4, 2, 2, 2><<<dim3(1, N_TOK / 128), blk, 0, stream>>>(
                H16, eW2h + (long)e * EXP_HID * MOE_DIM, e_b2 + (long)e * MOE_DIM,
                nullptr, moeF, comb, e, N_TOK, MOE_DIM, EXP_HID);
        else
            gemm_kernel<0, 1, 4, 2, 2, 2><<<dim3(1, N_TOK / 128), blk, 0, stream>>>(
                H16, eW2h + (long)e * EXP_HID * MOE_DIM, e_b2 + (long)e * MOE_DIM,
                nullptr, moeF, comb, e, N_TOK, MOE_DIM, EXP_HID);
    }
    f2h(moeF, moe16, (long)N_TOK * MOE_DIM, stream);
    // 5) rates = sigmoid(moe @ c2s_w + c2s_b)           [8192 x 2048]
    gemm_kernel<1, 0, 2, 4, 4, 2><<<dim3(HIDDEN / 128, N_TOK / 128), blk, 0, stream>>>(
        moe16, c2sW16, c2s_b, rates16, nullptr, nullptr, 0, N_TOK, HIDDEN, MOE_DIM);
    // 6) dec = sigmoid(rates @ dec_w + dec_b)           [8192 x 1024]
    gemm_kernel<1, 0, 2, 4, 4, 2><<<dim3(D_MODEL / 128, N_TOK / 128), blk, 0, stream>>>(
        rates16, decW16, dec_b, dec16, nullptr, nullptr, 0, N_TOK, D_MODEL, HIDDEN);
    // 7) LayerNorm -> d_out (f32)
    layernorm_kernel<<<N_TOK, blk, 0, stream>>>(dec16, ln_g, ln_b, (float*)d_out, D_MODEL);
}
